// DirectNorm2d_75591424409916
// MI455X (gfx1250) — matrix operand order repaired and checked by reading, not compiled, not run on hardware
//
#include <hip/hip_runtime.h>
#include <hip/hip_bf16.h>

// ---------------------------------------------------------------------------
// DirectNorm2d on MI455X (gfx1250, wave32, WMMA)
//
//  1) instance-norm stats of `input`               (bandwidth)
//  2) 3x3 offset conv on `style` (reflect pad)     (small, 1.4 G MAC)
//  3) pack w_dc -> bf16, transposed to [co][k]     (tiny)
//  4) bilinear deform-sample -> bf16 im2col A[M][K]  M=16384, K=4608
//  5) GEMM  s = A x W via v_wmma_f32_16x16x32_bf16 (77 GFLOP), double-
//     buffered LDS software pipeline, s written into d_out
//  6) stats of s (+ b_dc folded into s_mean; variance unaffected)
//  7) finalize: out = std(s)*instnorm(input) + mean(s)   (overwrites d_out)
// ---------------------------------------------------------------------------

typedef __attribute__((ext_vector_type(16))) __bf16 v16bf;
typedef __attribute__((ext_vector_type(8)))  float  v8f;

#define B_   4
#define C_   512
#define SD_  512
#define H_   64
#define W_   64
#define HW_  4096          // 64*64
#define KK_  9             // 3x3 taps
#define KDIM 4608          // 9*512  (GEMM K)
#define MDIM 16384         // B*H*W  (GEMM M)
#define NDIM 512           // C out  (GEMM N)

__device__ __forceinline__ unsigned short f2bf(float f) {
    unsigned u = __builtin_bit_cast(unsigned, f);
    unsigned r = u + 0x7FFFu + ((u >> 16) & 1u);   // round-to-nearest-even
    return (unsigned short)(r >> 16);
}

// ---------------------------------------------------------------------------
// Kernel 1/6: per-(b,c) spatial mean + scale over 4096 elements.
// mode 0: out_b = rsqrt(pop_var + 1e-5)   (instance norm of input)
// mode 1: out_b = sqrt(sample_var + 1e-8) (stats of s, ddof=1)
// ---------------------------------------------------------------------------
__global__ void dn_stats_kernel(const float* __restrict__ X,
                                float* __restrict__ out_mean,
                                float* __restrict__ out_b, int mode) {
    const int bc = blockIdx.x;
    const float* p = X + ((size_t)bc << 12);
    float s = 0.f, ss = 0.f;
    for (int i = threadIdx.x; i < HW_; i += 256) {
        float v = p[i];
        s += v; ss += v * v;
    }
    __shared__ float sh1[256], sh2[256];
    sh1[threadIdx.x] = s; sh2[threadIdx.x] = ss;
    __syncthreads();
    for (int off = 128; off > 0; off >>= 1) {
        if (threadIdx.x < off) {
            sh1[threadIdx.x] += sh1[threadIdx.x + off];
            sh2[threadIdx.x] += sh2[threadIdx.x + off];
        }
        __syncthreads();
    }
    if (threadIdx.x == 0) {
        float mean = sh1[0] * (1.0f / HW_);
        out_mean[bc] = mean;
        if (mode == 0) {
            float var = sh2[0] * (1.0f / HW_) - mean * mean;
            out_b[bc] = rsqrtf(var + 1e-5f);
        } else {
            float var = (sh2[0] - (float)HW_ * mean * mean) * (1.0f / (HW_ - 1));
            out_b[bc] = sqrtf(var + 1e-8f);
        }
    }
}

// ---------------------------------------------------------------------------
// Kernel 2: offset conv. One thread per (b,h,w); 18 accumulators over
// c=0..511 and 9 reflect-padded taps. Weight loads are wave-uniform
// (scalar path). off layout: [b][o][h][w], o = kk*2 + {dy=0,dx=1}.
// ---------------------------------------------------------------------------
__global__ void dn_offset_conv(const float* __restrict__ style,
                               const float* __restrict__ w_off,
                               const float* __restrict__ b_off,
                               float* __restrict__ off) {
    const int t = blockIdx.x * 256 + threadIdx.x;       // 0..16383
    const int w = t & 63, h = (t >> 6) & 63, b = t >> 12;

    float acc[18];
#pragma unroll
    for (int o = 0; o < 18; ++o) acc[o] = b_off[o];

    int yy[3], xx[3];
#pragma unroll
    for (int i = 0; i < 3; ++i) {
        int y = h - 1 + i; yy[i] = (y < 0) ? -y : ((y > 63) ? 126 - y : y);
        int x = w - 1 + i; xx[i] = (x < 0) ? -x : ((x > 63) ? 126 - x : x);
    }
    const float* sb = style + (size_t)b * (SD_ * HW_);
    for (int c = 0; c < SD_; ++c) {
        const float* sc = sb + ((size_t)c << 12);
        const float* wc = w_off + c * 9;                // w_off[o][c][t]: + o*4608
#pragma unroll
        for (int ky = 0; ky < 3; ++ky)
#pragma unroll
            for (int kx = 0; kx < 3; ++kx) {
                float v = sc[yy[ky] * 64 + xx[kx]];
                int ti = ky * 3 + kx;
#pragma unroll
                for (int o = 0; o < 18; ++o)
                    acc[o] = fmaf(v, wc[(size_t)o * KDIM + ti], acc[o]);
            }
    }
#pragma unroll
    for (int o = 0; o < 18; ++o)
        off[(((size_t)b * 18 + o) << 12) + (h << 6) + w] = acc[o];
}

// ---------------------------------------------------------------------------
// Kernel 3: pack w_dc (C,SD,3,3) -> bf16 B^T[co][k], k = kk*512 + c.
// ---------------------------------------------------------------------------
__global__ void dn_pack_w(const float* __restrict__ w_dc,
                          unsigned short* __restrict__ Bt) {
    int i = blockIdx.x * 256 + threadIdx.x;
    if (i >= NDIM * KDIM) return;
    int co = i / KDIM, k = i - co * KDIM;
    int kk = k >> 9, c = k & 511;
    Bt[i] = f2bf(w_dc[(size_t)co * KDIM + c * 9 + kk]);
}

// ---------------------------------------------------------------------------
// Kernel 4: bilinear deform-sampling -> bf16 im2col A[m][kk*512+c].
// One thread per position m=(b,h,w); taps/weights fixed across c.
// ---------------------------------------------------------------------------
__global__ void dn_sample_im2col(const float* __restrict__ style,
                                 const float* __restrict__ off,
                                 unsigned short* __restrict__ A) {
    const int t = blockIdx.x * 256 + threadIdx.x;       // m
    const int w = t & 63, h = (t >> 6) & 63, b = t >> 12;
    const float* sb = style + (size_t)b * (SD_ * HW_);
    unsigned short* arow = A + (size_t)t * KDIM;

    for (int kk = 0; kk < KK_; ++kk) {
        const int ky = kk / 3, kx = kk - 3 * ky;
        const float dy = off[(((size_t)b * 18 + kk * 2 + 0) << 12) + (h << 6) + w];
        const float dx = off[(((size_t)b * 18 + kk * 2 + 1) << 12) + (h << 6) + w];
        const float y = (float)(h - 1 + ky) + dy;
        const float x = (float)(w - 1 + kx) + dx;
        const float y0f = floorf(y), x0f = floorf(x);
        const float fy = y - y0f, fx = x - x0f;
        const int iy0 = (int)y0f, ix0 = (int)x0f;

        float wgt[4]; int idx[4];
#pragma unroll
        for (int tap = 0; tap < 4; ++tap) {
            const int oy = tap >> 1, ox = tap & 1;
            const int iy = iy0 + oy, ix = ix0 + ox;
            const bool valid = (iy >= 0) & (iy < H_) & (ix >= 0) & (ix < W_);
            const int iyc = min(max(iy, 0), H_ - 1);
            const int ixc = min(max(ix, 0), W_ - 1);
            const float wy = oy ? fy : (1.0f - fy);
            const float wx = ox ? fx : (1.0f - fx);
            wgt[tap] = valid ? wy * wx : 0.0f;
            idx[tap] = iyc * 64 + ixc;
        }
        unsigned short* ak = arow + kk * SD_;
        for (int c = 0; c < SD_; ++c) {
            const float* sc = sb + ((size_t)c << 12);
            float v = wgt[0] * sc[idx[0]] + wgt[1] * sc[idx[1]] +
                      wgt[2] * sc[idx[2]] + wgt[3] * sc[idx[3]];
            ak[c] = f2bf(v);
        }
    }
}

// ---------------------------------------------------------------------------
// Kernel 5: tiled bf16 WMMA GEMM, double-buffered LDS software pipeline.
// S[m][n] = sum_k A[m][k] * Bt[n][k].  Block tile 128x128, Ktile 32,
// 8 waves (4 along M x 2 along N), each wave 32x64 = 8 accumulators.
// Steady-state iteration: 4 global_load_b128 (next tile) -> frag ds_loads
// -> 8 v_wmma -> 4 ds_store_b128 -> 1 barrier.
// Output written as s[b][n][hw] (m = b*4096 + hw) into d_out.
// ---------------------------------------------------------------------------
__global__ __launch_bounds__(256)
void dn_gemm_wmma(const unsigned short* __restrict__ A,
                  const unsigned short* __restrict__ Bt,
                  float* __restrict__ S) {
    __shared__ unsigned short As[2][128 * 32];   // [buf][row][k] bf16, 2x8 KB
    __shared__ unsigned short Bs[2][128 * 32];   // [buf][n][k]   bf16, 2x8 KB

    const int tid  = threadIdx.x;
    const int lane = tid & 31;
    const int wv   = tid >> 5;
    const int wm   = wv & 3;          // wave row   (4 -> 32 M each)
    const int wn   = wv >> 2;         // wave col   (2 -> 64 N each)
    const int lrow = lane & 15;
    const int hi   = lane >> 4;       // lane half select

    const int m0 = blockIdx.x * 128;
    const int n0 = blockIdx.y * 128;

    const unsigned short* Ag = A  + (size_t)m0 * KDIM;
    const unsigned short* Bg = Bt + (size_t)n0 * KDIM;

    // staging geometry: 128 rows x 64B per matrix tile; 2 b128 segs/thread
    const int s0 = tid, s1 = tid + 256;
    const size_t g0 = (size_t)(s0 >> 2) * KDIM + (s0 & 3) * 8;
    const size_t g1 = (size_t)(s1 >> 2) * KDIM + (s1 & 3) * 8;
    const int    l0 = (s0 >> 2) * 32 + (s0 & 3) * 8;
    const int    l1 = (s1 >> 2) * 32 + (s1 & 3) * 8;

    v8f acc[2][4];
#pragma unroll
    for (int mi = 0; mi < 2; ++mi)
#pragma unroll
        for (int ni = 0; ni < 4; ++ni) acc[mi][ni] = (v8f)(0.0f);

    union Frag { v16bf v; uint4 q[2]; };

    // frag LDS offsets (elements) within a buffer
    const int aoff0 = (wm * 32 + 0 * 16 + lrow) * 32 + hi * 8;
    const int aoff1 = (wm * 32 + 1 * 16 + lrow) * 32 + hi * 8;
    const int boffb = (wn * 64 + lrow) * 32 + hi * 8;

    // ---- prologue: stage tile 0 into buffer 0
    *(uint4*)(&As[0][l0]) = *(const uint4*)(Ag + g0);
    *(uint4*)(&As[0][l1]) = *(const uint4*)(Ag + g1);
    *(uint4*)(&Bs[0][l0]) = *(const uint4*)(Bg + g0);
    *(uint4*)(&Bs[0][l1]) = *(const uint4*)(Bg + g1);
    __syncthreads();

    const int NIT = KDIM / 32;                 // 144
    for (int it = 0; it < NIT - 1; ++it) {
        const int cur = it & 1, nxt = cur ^ 1;
        const int kn = (it + 1) * 32;

        // stage next tile global->regs (overlaps with WMMA below)
        uint4 ra0 = *(const uint4*)(Ag + g0 + kn);
        uint4 ra1 = *(const uint4*)(Ag + g1 + kn);
        uint4 rb0 = *(const uint4*)(Bg + g0 + kn);
        uint4 rb1 = *(const uint4*)(Bg + g1 + kn);
        __builtin_prefetch(Ag + g0 + kn + 32, 0, 1);   // 2 tiles ahead
        __builtin_prefetch(Bg + g0 + kn + 32, 0, 1);

        // A fragments (16x32): VGPR0-3 = K(hi*8..+7), VGPR4-7 = K(16+hi*8..)
        Frag a[2];
        a[0].q[0] = *(const uint4*)(&As[cur][aoff0]);
        a[0].q[1] = *(const uint4*)(&As[cur][aoff0 + 16]);
        a[1].q[0] = *(const uint4*)(&As[cur][aoff1]);
        a[1].q[1] = *(const uint4*)(&As[cur][aoff1 + 16]);
        // B fragments (32x16): VGPR0-3 = K(hi*8..+7), VGPR4-7 = K(16+hi*8..)
        Frag bfr[4];
#pragma unroll
        for (int ni = 0; ni < 4; ++ni) {
            bfr[ni].q[0] = *(const uint4*)(&Bs[cur][boffb + ni * 512]);
            bfr[ni].q[1] = *(const uint4*)(&Bs[cur][boffb + ni * 512 + 16]);
        }
#pragma unroll
        for (int mi = 0; mi < 2; ++mi)
#pragma unroll
            for (int ni = 0; ni < 4; ++ni)
                acc[mi][ni] = __builtin_amdgcn_wmma_f32_16x16x32_bf16(
                    false, a[mi].v, false, bfr[ni].v,
                    (short)0, acc[mi][ni], false, false);

        // commit next tile to the other buffer
        *(uint4*)(&As[nxt][l0]) = ra0;
        *(uint4*)(&As[nxt][l1]) = ra1;
        *(uint4*)(&Bs[nxt][l0]) = rb0;
        *(uint4*)(&Bs[nxt][l1]) = rb1;
        __syncthreads();
    }

    // ---- epilogue iteration (tile NIT-1, buffer (NIT-1)&1)
    {
        const int cur = (NIT - 1) & 1;
        Frag a[2];
        a[0].q[0] = *(const uint4*)(&As[cur][aoff0]);
        a[0].q[1] = *(const uint4*)(&As[cur][aoff0 + 16]);
        a[1].q[0] = *(const uint4*)(&As[cur][aoff1]);
        a[1].q[1] = *(const uint4*)(&As[cur][aoff1 + 16]);
        Frag bfr[4];
#pragma unroll
        for (int ni = 0; ni < 4; ++ni) {
            bfr[ni].q[0] = *(const uint4*)(&Bs[cur][boffb + ni * 512]);
            bfr[ni].q[1] = *(const uint4*)(&Bs[cur][boffb + ni * 512 + 16]);
        }
#pragma unroll
        for (int mi = 0; mi < 2; ++mi)
#pragma unroll
            for (int ni = 0; ni < 4; ++ni)
                acc[mi][ni] = __builtin_amdgcn_wmma_f32_16x16x32_bf16(
                    false, a[mi].v, false, bfr[ni].v,
                    (short)0, acc[mi][ni], false, false);
    }

    // Epilogue: VGPR r -> M = m0+wm*32+mi*16+8*hi + r, N = lrow; the 8 rows
    // per lane are consecutive hw -> pack into two float4 stores each.
#pragma unroll
    for (int mi = 0; mi < 2; ++mi) {
        const int m  = m0 + wm * 32 + mi * 16 + 8 * hi;   // 8 consecutive rows
        const int b  = m >> 12;
        const int hw = m & 4095;
#pragma unroll
        for (int ni = 0; ni < 4; ++ni) {
            const int n = n0 + wn * 64 + ni * 16 + lrow;
            float* dst = S + (((size_t)b * NDIM + n) << 12) + hw;
            float4 v0 = make_float4(acc[mi][ni][0], acc[mi][ni][1],
                                    acc[mi][ni][2], acc[mi][ni][3]);
            float4 v1 = make_float4(acc[mi][ni][4], acc[mi][ni][5],
                                    acc[mi][ni][6], acc[mi][ni][7]);
            *(float4*)(dst)     = v0;
            *(float4*)(dst + 4) = v1;
        }
    }
}

// ---------------------------------------------------------------------------
// Kernel 6b: fold b_dc into s_mean (b_dc shifts s uniformly over (h,w):
// mean gains b_dc, variance unchanged).
// ---------------------------------------------------------------------------
__global__ void dn_add_bias(float* __restrict__ s_mean,
                            const float* __restrict__ b_dc) {
    int i = blockIdx.x * 256 + threadIdx.x;
    if (i < B_ * C_) s_mean[i] += b_dc[i & (C_ - 1)];
}

// ---------------------------------------------------------------------------
// Kernel 7: out = sqrt(var_s)*instnorm(input) + mean_s   (float4 vectorized)
// ---------------------------------------------------------------------------
__global__ void dn_finalize(const float* __restrict__ input,
                            const float* __restrict__ mean_in,
                            const float* __restrict__ rstd_in,
                            const float* __restrict__ s_mean,
                            const float* __restrict__ s_std,
                            float* __restrict__ out) {
    const size_t i = (size_t)blockIdx.x * 256 + threadIdx.x;   // float4 index
    const int bc = (int)(i >> 10);                             // 1024 f4 / (b,c)
    const float m  = mean_in[bc], rs = rstd_in[bc];
    const float sm = s_mean[bc],  sd = s_std[bc];
    float4 v = ((const float4*)input)[i];
    v.x = sd * (v.x - m) * rs + sm;
    v.y = sd * (v.y - m) * rs + sm;
    v.z = sd * (v.z - m) * rs + sm;
    v.w = sd * (v.w - m) * rs + sm;
    ((float4*)out)[i] = v;
}

// ---------------------------------------------------------------------------
extern "C" void kernel_launch(void* const* d_in, const int* in_sizes, int n_in,
                              void* d_out, int out_size, void* d_ws, size_t ws_size,
                              hipStream_t stream) {
    const float* input = (const float*)d_in[0];
    const float* style = (const float*)d_in[1];
    const float* w_off = (const float*)d_in[2];
    const float* b_off = (const float*)d_in[3];
    const float* w_dc  = (const float*)d_in[4];
    const float* b_dc  = (const float*)d_in[5];

    char* ws = (char*)d_ws;
    float* mean_in = (float*)(ws + 0);                        //  8 KB
    float* rstd_in = (float*)(ws + 8192);                     //  8 KB
    float* s_mean  = (float*)(ws + 16384);                    //  8 KB
    float* s_std   = (float*)(ws + 24576);                    //  8 KB
    float* off     = (float*)(ws + 32768);                    // 1.13 MB
    unsigned short* Bt = (unsigned short*)(ws + 1212416);     // 4.5 MB
    unsigned short* A  = (unsigned short*)(ws + 5931008);     // 144 MB
    float* s = (float*)d_out;   // s lives in d_out; finalize overwrites it

    // 1) instance-norm stats of input
    dn_stats_kernel<<<dim3(B_ * C_), dim3(256), 0, stream>>>(input, mean_in, rstd_in, 0);
    // 2) offset conv
    dn_offset_conv<<<dim3(MDIM / 256), dim3(256), 0, stream>>>(style, w_off, b_off, off);
    // 3) pack weights (bf16, transposed)
    dn_pack_w<<<dim3((NDIM * KDIM + 255) / 256), dim3(256), 0, stream>>>(w_dc, Bt);
    // 4) deform sampling -> bf16 im2col
    dn_sample_im2col<<<dim3(MDIM / 256), dim3(256), 0, stream>>>(style, off, A);
    // 5) WMMA GEMM -> s (in d_out)
    dn_gemm_wmma<<<dim3(MDIM / 128, NDIM / 128), dim3(256), 0, stream>>>(A, Bt, s);
    // 6) stats of s, then fold in b_dc
    dn_stats_kernel<<<dim3(B_ * C_), dim3(256), 0, stream>>>(s, s_mean, s_std, 1);
    dn_add_bias<<<dim3((B_ * C_ + 255) / 256), dim3(256), 0, stream>>>(s_mean, b_dc);
    // 7) finalize (overwrites d_out; reads only input + stats)
    dn_finalize<<<dim3((B_ * C_ * HW_ / 4) / 256), dim3(256), 0, stream>>>(
        input, mean_in, rstd_in, s_mean, s_std, (float*)d_out);
}
